// FusedDenseLayer_69879117906304
// MI455X (gfx1250) — matrix operand order repaired and checked by reading, not compiled, not run on hardware
//
#include <hip/hip_runtime.h>

typedef __bf16  v16bf __attribute__((ext_vector_type(16)));
typedef __bf16  v4bf  __attribute__((ext_vector_type(4)));
typedef float   v8f   __attribute__((ext_vector_type(8)));

#define BN_EPS   1e-5f
#define CIN      512
#define COUT     32
#define HH       56
#define WW       56
#define NPIX     3136          // 56*56
#define CBLK     32            // input channels per K-block (K=32 bf16 WMMA)
#define NCB      16            // 512 / 32
#define WPACK_HALFS 147456     // 2 halves * 9 taps * 16 cb * 32 lanes * 16

// 2-D output tile per workgroup: 16 wide x 8 high, 8 waves (one row each)
#define TLW      16
#define TLH      8
#define HCOLS    18            // TLW + 2 halo cols
#define HROWS    10            // TLH + 2 halo rows
#define HPIX     180           // HCOLS * HROWS staged halo pixels
#define NTX      4             // ceil(56/16) column tiles (last masked)
#define NTY      7             // 56/8 row tiles

// ---------------- kernel 0: fold BN into per-channel scale/shift ----------------
__global__ void bn_coef_kernel(const float* __restrict__ w, const float* __restrict__ b,
                               const float* __restrict__ mean, const float* __restrict__ var,
                               float* __restrict__ scale, float* __restrict__ shift) {
    int c = blockIdx.x * blockDim.x + threadIdx.x;
    if (c < CIN) {
        float s = w[c] * rsqrtf(var[c] + BN_EPS);
        scale[c] = s;
        shift[c] = b[c] - mean[c] * s;
    }
}

// ---------------- kernel 1: pack conv weights into WMMA A-layout (bf16) --------
// Packed A (16x32 MxK), contiguous K, not ISA table order: lanes 0-15 hold K=0..15, lanes 16-31 hold
// K=16..31; inside v16bf, K = 16*g + j (matches contiguous B from LDS).
__global__ void pack_weights_kernel(const float* __restrict__ cw, __bf16* __restrict__ wp) {
    int e = blockIdx.x * blockDim.x + threadIdx.x;
    if (e >= WPACK_HALFS) return;
    int j    = e & 15;
    int lane = (e >> 4) & 31;
    int tile = e >> 9;
    int half = tile & 1;
    int rest = tile >> 1;
    int t    = rest % 9;
    int cb   = rest / 9;
    int g    = lane >> 4;
    int M    = lane & 15;
    int K    = j + g * 16;
    int co   = half * 16 + M;
    int ci   = cb * CBLK + K;
    int kh   = t / 3, kw = t % 3;
    wp[e] = (__bf16)cw[((co * CIN + ci) * 3 + kh) * 3 + kw];
}

// ---------------- kernel 2: fused BN+ReLU -> implicit-GEMM 3x3 conv ------------
__global__ __launch_bounds__(256)
void conv_wmma_kernel(const float* __restrict__ x,
                      const __bf16* __restrict__ wp,
                      const float* __restrict__ scale,
                      const float* __restrict__ shift,
                      float* __restrict__ out) {
    __shared__ __align__(64) __bf16 tile[HPIX * CBLK];   // [haloPixel][channel]

    const int bid  = blockIdx.x;
    const int n    = bid / (NTX * NTY);
    const int rtt  = bid % (NTX * NTY);
    const int ty   = rtt >> 2;            // 0..6
    const int tx   = rtt & 3;             // 0..3
    const int h0   = ty * TLH;
    const int x0   = tx * TLW;

    const int tid  = threadIdx.x;
    const int wave = tid >> 5;            // 0..7 : tile row / fill channel group
    const int lane = tid & 31;
    const int g    = lane >> 4;           // K-half group for B / D row group
    const int col  = lane & 15;           // B/D column
    const int myh  = h0 + wave;           // output row (always < 56)
    const int mywc = x0 + col;            // output col (may be >= 56 on last tile)

    // ---- per-lane, K-invariant tap metadata: LDS element offset + validity ----
    int  boff[9];
    bool vmask[9];
    #pragma unroll
    for (int t = 0; t < 9; ++t) {
        const int dh = t / 3 - 1, dw = t % 3 - 1;
        const int hs = myh + dh, ws = mywc + dw;
        vmask[t] = ((unsigned)hs < (unsigned)HH) && ((unsigned)ws < (unsigned)WW);
        const int pixIdx = (wave + dh + 1) * HCOLS + (col + dw + 1);
        boff[t] = pixIdx * CBLK + g * 16;
    }

    v8f acc0 = {};   // output channels 0..15
    v8f acc1 = {};   // output channels 16..31

    // running A pointer: walks packed weights linearly over (cb, tap)
    const v16bf* ap = reinterpret_cast<const v16bf*>(wp) + lane;

    for (int cb = 0; cb < NCB; ++cb) {
        __syncthreads();
        // ---- cooperative LDS fill: BN + ReLU + f32->bf16 ----
        // wave w owns channels 4w..4w+3 (scale/shift loop-invariant);
        // lanes walk consecutive halo pixels -> coalesced global reads;
        // 4 channels pack into one ds_store_b64.
        {
            const int c0 = cb * CBLK + wave * 4;
            const float sc0 = scale[c0],     sh0 = shift[c0];
            const float sc1 = scale[c0 + 1], sh1 = shift[c0 + 1];
            const float sc2 = scale[c0 + 2], sh2 = shift[c0 + 2];
            const float sc3 = scale[c0 + 3], sh3 = shift[c0 + 3];
            const float* xb = x + ((size_t)n * CIN + c0) * NPIX;

            #pragma unroll
            for (int k = 0; k < 6; ++k) {
                const int pix = lane + 32 * k;
                if (pix < HPIX) {
                    const int row  = pix / HCOLS;
                    const int colI = pix - row * HCOLS;
                    const int gh   = h0 - 1 + row;
                    const int gw   = x0 - 1 + colI;
                    const bool inb = ((unsigned)gh < (unsigned)HH) &&
                                     ((unsigned)gw < (unsigned)WW);
                    const int off  = gh * WW + gw;
                    if (cb + 1 < NCB && k == 0) {
                        int po = inb ? off : 0;
                        __builtin_prefetch(xb + (size_t)CBLK * NPIX + po, 0, 0);
                    }
                    float v0 = 0.f, v1 = 0.f, v2 = 0.f, v3 = 0.f;
                    if (inb) {
                        v0 = fmaxf(fmaf(xb[off],                     sc0, sh0), 0.f);
                        v1 = fmaxf(fmaf(xb[off + (size_t)NPIX],      sc1, sh1), 0.f);
                        v2 = fmaxf(fmaf(xb[off + (size_t)2 * NPIX],  sc2, sh2), 0.f);
                        v3 = fmaxf(fmaf(xb[off + (size_t)3 * NPIX],  sc3, sh3), 0.f);
                    }
                    v4bf pk;
                    pk.x = (__bf16)v0; pk.y = (__bf16)v1;
                    pk.z = (__bf16)v2; pk.w = (__bf16)v3;
                    *reinterpret_cast<v4bf*>(&tile[pix * CBLK + wave * 4]) = pk;
                }
            }
        }
        __syncthreads();

        #pragma unroll
        for (int t = 0; t < 9; ++t) {
            v16bf bmat = {};
            if (vmask[t])
                bmat = *reinterpret_cast<const v16bf*>(&tile[boff[t]]);

            v16bf a0 = ap[0];
            v16bf a1 = ap[32];
            ap += 64;

            acc0 = __builtin_amdgcn_wmma_f32_16x16x32_bf16(
                       false, a0, false, bmat, (short)0, acc0, false, false);
            acc1 = __builtin_amdgcn_wmma_f32_16x16x32_bf16(
                       false, a1, false, bmat, (short)0, acc1, false, false);
        }
    }

    // D layout: lanes 0-15 VGPR r -> M=r ; lanes 16-31 VGPR r -> M=8+r
    if (mywc < WW) {
        float* ob = out + (size_t)n * COUT * NPIX + myh * WW + mywc;
        #pragma unroll
        for (int r = 0; r < 8; ++r) {
            int m = g * 8 + r;
            ob[(size_t)m * NPIX]        = acc0[r];
            ob[(size_t)(16 + m) * NPIX] = acc1[r];
        }
    }
}

extern "C" void kernel_launch(void* const* d_in, const int* in_sizes, int n_in,
                              void* d_out, int out_size, void* d_ws, size_t ws_size,
                              hipStream_t stream) {
    (void)in_sizes; (void)n_in; (void)out_size; (void)ws_size;
    const float* x     = (const float*)d_in[0];
    // d_in[1] = out_map (zeros; reference returns just the conv slice -> unused)
    const float* bnw   = (const float*)d_in[2];
    const float* bnb   = (const float*)d_in[3];
    const float* rmean = (const float*)d_in[4];
    const float* rvar  = (const float*)d_in[5];
    const float* cw    = (const float*)d_in[6];
    // d_in[7] = write_offset (cancels out in the reference) -> unused

    __bf16* wp   = (__bf16*)d_ws;
    float* scale = (float*)((char*)d_ws + (size_t)WPACK_HALFS * 2);
    float* shift = scale + CIN;
    float* out   = (float*)d_out;

    bn_coef_kernel<<<2, 256, 0, stream>>>(bnw, bnb, rmean, rvar, scale, shift);
    pack_weights_kernel<<<WPACK_HALFS / 256, 256, 0, stream>>>(cw, wp);
    conv_wmma_kernel<<<32 * NTX * NTY, 256, 0, stream>>>(x, wp, scale, shift, out);
}
